// AttentionHead_8787503087617
// MI455X (gfx1250) — matrix operand order repaired and checked by reading, not compiled, not run on hardware
//
#include <hip/hip_runtime.h>
#include <hip/hip_bf16.h>

#define B_ 8
#define T_ 2048
#define C_ 1024
#define H_ 64

typedef __attribute__((ext_vector_type(16))) _Float16 v16h;
typedef __attribute__((ext_vector_type(8)))  _Float16 v8h;
typedef __attribute__((ext_vector_type(8)))  float    v8f;
typedef __attribute__((ext_vector_type(4)))  unsigned int u32x4;
typedef __attribute__((ext_vector_type(4)))  int      i32x4;
typedef __attribute__((ext_vector_type(8)))  int      i32x8;

#if defined(__has_builtin)
#if __has_builtin(__builtin_amdgcn_tensor_load_to_lds)
#define HAVE_TDM 1
#endif
#endif

static __device__ __forceinline__ v16h cat8(v8h lo, v8h hi) {
  return __builtin_shufflevector(lo, hi, 0,1,2,3,4,5,6,7,8,9,10,11,12,13,14,15);
}

static __device__ __forceinline__ v8f wmma_f16(v16h a, v16h b, v8f c) {
  return __builtin_amdgcn_wmma_f32_16x16x32_f16(false, a, false, b, (short)0, c, false, false);
}

// ---------------- conversion kernels ----------------

__global__ void cvt_x_kernel(const float* __restrict__ x, _Float16* __restrict__ xh, int n) {
  int i = blockIdx.x * 256 + threadIdx.x;
  if (i < n) xh[i] = (_Float16)x[i];
}

// W [C,H] f32 -> Wt [H,C] f16 (per z in {q,k,v})
__global__ void cvt_w_kernel(const float* __restrict__ Wq, const float* __restrict__ Wk,
                             const float* __restrict__ Wv, _Float16* __restrict__ Wt) {
  int i = blockIdx.x * 256 + threadIdx.x;       // 0 .. C*H-1
  int z = blockIdx.y;
  const float* W = (z == 0) ? Wq : (z == 1) ? Wk : Wv;
  int k = i / H_;
  int n = i % H_;
  Wt[(size_t)z * C_ * H_ + (size_t)n * C_ + k] = (_Float16)W[i];
}

// ---------------- fused QKV projection GEMM ----------------
// grid: (B*T/16). block: 128 (4 waves). Block stages its 16x1024 f16 x-panel
// into LDS once (TDM if available), then wave w computes the 16x16 tiles at
// cols [w*16,+16) for ALL of Q, K, V in one K-loop (A fragment reused 3x).
// Q,K stored [B*T][H] f16; V stored transposed [B][H][T] f16.
__global__ __launch_bounds__(128)
void qkv_proj_kernel(const _Float16* __restrict__ xh, const _Float16* __restrict__ Wt,
                     const float* __restrict__ bq, const float* __restrict__ bk,
                     const float* __restrict__ bv,
                     _Float16* __restrict__ Qh, _Float16* __restrict__ Kh,
                     _Float16* __restrict__ Vt)
{
  __shared__ _Float16 Atile[16 * C_];           // 32 KB x-panel

  const int lane  = threadIdx.x & 31;
  const int wave  = threadIdx.x >> 5;
  const int mbase = blockIdx.x * 16;
  const int n     = lane & 15;
  const int hsel  = lane >> 4;
  const int nglob = wave * 16 + n;

  // ---- stage x panel: global -> LDS ----
#if defined(HAVE_TDM)
  if (wave == 0) {
    const unsigned long long ga = (unsigned long long)(size_t)(xh + (size_t)mbase * C_);
    const unsigned int ldsa = (unsigned int)(size_t)&Atile[0];
    // D# group0: count=1 | lds_addr | global_addr[56:0] | type=2 ("image")
    u32x4 g0;
    g0[0] = 1u;
    g0[1] = ldsa;
    g0[2] = (unsigned int)(ga & 0xFFFFFFFFull);
    g0[3] = (unsigned int)((ga >> 32) & 0x01FFFFFFull) | (2u << 30);
    // D# group1: 1-D tile of 16*1024 = 16384 2-byte elements
    // w0: data_size=1 (2B) at bits[17:16]; w1[31:16]=tensor_dim0 lo;
    // w2[31:16]=tensor_dim1 lo (=1); w3[31:16]=tile_dim0; w5=tensor_dim0_stride
    i32x8 g1;
    g1[0] = 0x00010000;
    g1[1] = (int)(0x4000u << 16);
    g1[2] = (int)(1u << 16);
    g1[3] = (int)(0x4000u << 16);
    g1[4] = 0;
    g1[5] = 0x4000;
    g1[6] = 0;
    g1[7] = 0;
    i32x4 g2 = {0, 0, 0, 0};
    i32x4 g3 = {0, 0, 0, 0};
#if __clang_major__ >= 23
    i32x8 gx = {0, 0, 0, 0, 0, 0, 0, 0};
    __builtin_amdgcn_tensor_load_to_lds(g0, g1, g2, g3, gx, 0);
#else
    __builtin_amdgcn_tensor_load_to_lds(g0, g1, g2, g3, 0);
#endif
    __builtin_amdgcn_s_wait_tensorcnt(0);
  }
#else
  // fallback: cooperative 128-thread copy (16B per thread per step)
  for (int i = threadIdx.x; i < 16 * C_ / 8; i += 128)
    *(v8h*)&Atile[i * 8] = *(const v8h*)(xh + (size_t)mbase * C_ + (size_t)i * 8);
#endif
  __syncthreads();

  const _Float16* bq_row = Wt + 0 * (size_t)C_ * H_ + (size_t)nglob * C_;
  const _Float16* bk_row = Wt + 1 * (size_t)C_ * H_ + (size_t)nglob * C_;
  const _Float16* bv_row = Wt + 2 * (size_t)C_ * H_ + (size_t)nglob * C_;
  const _Float16* arow   = &Atile[n * C_];      // A operand: row m = lane&15 (from LDS)

  v8f accq = {}, acck = {}, accv = {};
  #pragma unroll 4
  for (int k = 0; k < C_; k += 32) {
    // A 16x32 f16: per-lane 16 contiguous K values (koff = 0|16 by lane half)
    v16h a = cat8(*(const v8h*)(arow + k + hsel * 16),
                  *(const v8h*)(arow + k + hsel * 16 + 8));
    // B 32x16 f16: per-lane 16 contiguous K values (koff = 0|16 by lane half)
    v16h wq = cat8(*(const v8h*)(bq_row + k + hsel * 16),
                   *(const v8h*)(bq_row + k + hsel * 16 + 8));
    accq = wmma_f16(a, wq, accq);
    v16h wk = cat8(*(const v8h*)(bk_row + k + hsel * 16),
                   *(const v8h*)(bk_row + k + hsel * 16 + 8));
    acck = wmma_f16(a, wk, acck);
    v16h wv = cat8(*(const v8h*)(bv_row + k + hsel * 16),
                   *(const v8h*)(bv_row + k + hsel * 16 + 8));
    accv = wmma_f16(a, wv, accv);
  }

  const float biasq = bq[nglob];
  const float biask = bk[nglob];
  const float biasv = bv[nglob];
  #pragma unroll
  for (int r = 0; r < 8; ++r) { accq[r] += biasq; acck[r] += biask; accv[r] += biasv; }

  // Q, K: row-major [t][h] f16 (strided per-element stores; D layout row = VGPR idx)
  #pragma unroll
  for (int r = 0; r < 8; ++r) {
    int m = mbase + r + hsel * 8;
    Qh[(size_t)m * H_ + nglob] = (_Float16)accq[r];
    Kh[(size_t)m * H_ + nglob] = (_Float16)acck[r];
  }
  // V transposed [b][h][t]: lane's 8 values are 8 consecutive t -> one 16B store
  {
    int b  = mbase / T_;
    int t0 = mbase - b * T_ + hsel * 8;
    v8h hv;
    #pragma unroll
    for (int r = 0; r < 8; ++r) hv[r] = (_Float16)accv[r];
    *(v8h*)(Vt + ((size_t)(b * H_ + nglob)) * T_ + t0) = hv;
  }
}

// ---------------- flash attention ----------------
// grid: (T/64, B). block: 128 (4 waves). Wave handles one 16-row query tile:
// qt = blockIdx.x*4 + wave. Key blocks of 32 with online softmax.
__global__ __launch_bounds__(128)
void flash_attn_kernel(const _Float16* __restrict__ Qh, const _Float16* __restrict__ Kh,
                       const _Float16* __restrict__ Vt, float* __restrict__ out)
{
  __shared__ _Float16 Plds[4][16 * 32];   // per-wave P tile (16 rows x 32 keys)

  const int lane  = threadIdx.x & 31;
  const int wave  = threadIdx.x >> 5;
  const int b     = blockIdx.y;
  const int qt    = blockIdx.x * 4 + wave;   // 0..127
  const int qbase = qt * 16;
  const int n     = lane & 15;
  const int hsel  = lane >> 4;

  // Q A-operands for h chunks [0,32) and [32,64), loaded once
  const _Float16* qrow = Qh + ((size_t)(b * T_ + qbase + n)) * H_;
  const v16h aQ0 = cat8(*(const v8h*)(qrow + hsel * 16),
                        *(const v8h*)(qrow + hsel * 16 + 8));
  const v16h aQ1 = cat8(*(const v8h*)(qrow + 32 + hsel * 16),
                        *(const v8h*)(qrow + 32 + hsel * 16 + 8));

  v8f o[4];
  #pragma unroll
  for (int i = 0; i < 4; ++i) o[i] = (v8f){};
  float mrow[8], lrow[8];
  #pragma unroll
  for (int r = 0; r < 8; ++r) { mrow[r] = -3.0e38f; lrow[r] = 0.0f; }

  const float scale = 0.03125f;            // 1/sqrt(C)
  const int nblocks = qt / 2 + 1;          // causal: keys 0 .. qbase+15
  const _Float16* Kbase = Kh + ((size_t)b * T_) * H_;
  const _Float16* Vbase = Vt + ((size_t)b * H_) * T_;
  _Float16* Pw = &Plds[wave][0];

  for (int kb = 0; kb < nblocks; ++kb) {
    const int sbase = kb * 32;

    // ---- S = Q K^T for keys [sbase, sbase+32) ----
    const _Float16* k0p = Kbase + ((size_t)(sbase + n)) * H_;
    const _Float16* k1p = Kbase + ((size_t)(sbase + 16 + n)) * H_;
    if (kb + 1 < nblocks) {
      __builtin_prefetch(k0p + 32 * H_, 0, 3);   // next key block (K rows)
      __builtin_prefetch(k1p + 32 * H_, 0, 3);
    }
    v16h b00 = cat8(*(const v8h*)(k0p + hsel * 16),      *(const v8h*)(k0p + hsel * 16 + 8));
    v16h b01 = cat8(*(const v8h*)(k0p + 32 + hsel * 16), *(const v8h*)(k0p + 32 + hsel * 16 + 8));
    v16h b10 = cat8(*(const v8h*)(k1p + hsel * 16),      *(const v8h*)(k1p + hsel * 16 + 8));
    v16h b11 = cat8(*(const v8h*)(k1p + 32 + hsel * 16), *(const v8h*)(k1p + 32 + hsel * 16 + 8));

    v8f s0 = {}, s1 = {};
    s0 = wmma_f16(aQ0, b00, s0);
    s0 = wmma_f16(aQ1, b01, s0);
    s1 = wmma_f16(aQ0, b10, s1);
    s1 = wmma_f16(aQ1, b11, s1);

    // ---- scale + causal mask (C/D layout: VGPR r -> row, lane -> col) ----
    float e0[8], e1[8];
    #pragma unroll
    for (int r = 0; r < 8; ++r) {
      int mg  = qbase + r + hsel * 8;
      int sg0 = sbase + n;
      int sg1 = sbase + 16 + n;
      e0[r] = (sg0 <= mg) ? s0[r] * scale : -3.0e38f;
      e1[r] = (sg1 <= mg) ? s1[r] * scale : -3.0e38f;
    }

    // ---- online softmax update ----
    #pragma unroll
    for (int r = 0; r < 8; ++r) {
      float tm = fmaxf(e0[r], e1[r]);
      tm = fmaxf(tm, __shfl_xor(tm, 1, 32));
      tm = fmaxf(tm, __shfl_xor(tm, 2, 32));
      tm = fmaxf(tm, __shfl_xor(tm, 4, 32));
      tm = fmaxf(tm, __shfl_xor(tm, 8, 32));   // masks <16: stays within lane half = row group
      float mnew = fmaxf(mrow[r], tm);
      float corr = __expf(mrow[r] - mnew);
      mrow[r] = mnew;
      float p0 = __expf(e0[r] - mnew);
      float p1 = __expf(e1[r] - mnew);
      float rs = p0 + p1;
      rs += __shfl_xor(rs, 1, 32);
      rs += __shfl_xor(rs, 2, 32);
      rs += __shfl_xor(rs, 4, 32);
      rs += __shfl_xor(rs, 8, 32);
      lrow[r] = lrow[r] * corr + rs;
      o[0][r] *= corr;
      o[1][r] *= corr;
      o[2][r] *= corr;
      o[3][r] *= corr;
      // stash P (C layout) into LDS row-major [row][s]
      Pw[(r + hsel * 8) * 32 + n]      = (_Float16)p0;
      Pw[(r + hsel * 8) * 32 + 16 + n] = (_Float16)p1;
    }

    // ---- reload P in A layout: per-lane row = lane&15, 16 contiguous s at 0|16 ----
    const _Float16* prow = Pw + (size_t)n * 32 + hsel * 16;
    v16h aP = cat8(*(const v8h*)prow, *(const v8h*)(prow + 8));

    // ---- O += P @ V  (V transposed [h][t]: per-lane 16 contiguous s) ----
    #pragma unroll
    for (int ht = 0; ht < 4; ++ht) {
      const _Float16* vp = Vbase + ((size_t)(ht * 16 + n)) * T_ + sbase + hsel * 16;
      v16h bV = cat8(*(const v8h*)vp, *(const v8h*)(vp + 8));
      o[ht] = wmma_f16(aP, bV, o[ht]);
    }
  }

  // ---- epilogue: normalize and store fp32 ----
  #pragma unroll
  for (int r = 0; r < 8; ++r) {
    float inv = 1.0f / lrow[r];
    size_t m = (size_t)(b * T_ + qbase + r + hsel * 8);
    #pragma unroll
    for (int ht = 0; ht < 4; ++ht)
      out[m * H_ + ht * 16 + n] = o[ht][r] * inv;
  }
}

// ---------------- launcher ----------------
extern "C" void kernel_launch(void* const* d_in, const int* in_sizes, int n_in,
                              void* d_out, int out_size, void* d_ws, size_t ws_size,
                              hipStream_t stream) {
  (void)in_sizes; (void)n_in; (void)out_size; (void)ws_size;
  const float* x  = (const float*)d_in[0];
  const float* Wq = (const float*)d_in[1];
  const float* bq = (const float*)d_in[2];
  const float* Wk = (const float*)d_in[3];
  const float* bk = (const float*)d_in[4];
  const float* Wv = (const float*)d_in[5];
  const float* bv = (const float*)d_in[6];
  float* out = (float*)d_out;

  char* ws = (char*)d_ws;
  size_t off = 0;
  _Float16* xh = (_Float16*)(ws + off); off += (size_t)B_ * T_ * C_ * 2;   // 32 MB
  _Float16* Qh = (_Float16*)(ws + off); off += (size_t)B_ * T_ * H_ * 2;   // 2 MB
  _Float16* Kh = (_Float16*)(ws + off); off += (size_t)B_ * T_ * H_ * 2;   // 2 MB
  _Float16* Vt = (_Float16*)(ws + off); off += (size_t)B_ * T_ * H_ * 2;   // 2 MB
  _Float16* Wt = (_Float16*)(ws + off); off += (size_t)3 * C_ * H_ * 2;    // 384 KB

  const int nx = B_ * T_ * C_;
  cvt_x_kernel<<<nx / 256, 256, 0, stream>>>(x, xh, nx);

  dim3 gw(C_ * H_ / 256, 3);
  cvt_w_kernel<<<gw, 256, 0, stream>>>(Wq, Wk, Wv, Wt);

  qkv_proj_kernel<<<B_ * T_ / 16, 128, 0, stream>>>(xh, Wt, bq, bk, bv, Qh, Kh, Vt);

  dim3 ga(T_ / 64, B_);
  flash_attn_kernel<<<ga, 128, 0, stream>>>(Qh, Kh, Vt, out);
}
